// SpikingLayer_29970281791985
// MI455X (gfx1250) — matrix operand order repaired and checked by reading, not compiled, not run on hardware
//
#include <hip/hip_runtime.h>
#include <hip/hip_bf16.h>
#include <math.h>

// Problem dims
#define B_  16
#define S_  2048
#define H_  1024
#define G_  4096          // 4*H
#define BS_ (B_*S_)       // 32768

typedef __attribute__((ext_vector_type(16))) __bf16 v16bf;
typedef __attribute__((ext_vector_type(8)))  __bf16 v8bf;
typedef __attribute__((ext_vector_type(8)))  float  v8f;
// GCC-style vectors to match the async/TDM builtin parameter types
typedef int          v4i __attribute__((vector_size(16)));
typedef int          v8i __attribute__((vector_size(32)));
typedef unsigned int v4u __attribute__((vector_size(16)));

#define AS1 __attribute__((address_space(1)))
#define AS3 __attribute__((address_space(3)))

#ifndef __has_builtin
#define __has_builtin(x) 0
#endif

#if __has_builtin(__builtin_amdgcn_global_load_async_to_lds_b128)
#define HAVE_ASYNC_LDS 1
#else
#define HAVE_ASYNC_LDS 0
#endif

#if __has_builtin(__builtin_amdgcn_tensor_load_to_lds)
#define HAVE_TDM 1
#else
#define HAVE_TDM 0
#endif

__device__ __forceinline__ v8f wmma_bf16(v16bf a, v16bf b, v8f c) {
  return __builtin_amdgcn_wmma_f32_16x16x32_bf16(false, a, false, b, (short)0, c, false, false);
}
__device__ __forceinline__ float sigmoidf_(float x) { return 1.0f / (1.0f + __expf(-x)); }

// ---- async copy 16B global -> LDS (ASYNCcnt path; falls back to ld+st) ----
__device__ __forceinline__ void cp16_g2l(const __bf16* g, __bf16* l) {
#if HAVE_ASYNC_LDS
  __builtin_amdgcn_global_load_async_to_lds_b128(
      (AS1 v4i*)(AS1 void*)(void*)g,
      (AS3 v4i*)(AS3 void*)(void*)l, 0, 0);
#else
  *(v8bf*)l = *(const v8bf*)g;
#endif
}
__device__ __forceinline__ void wait_async0() {
#if HAVE_ASYNC_LDS
#if __has_builtin(__builtin_amdgcn_s_wait_asynccnt)
  __builtin_amdgcn_s_wait_asynccnt(0);
#else
  asm volatile("s_wait_asynccnt 0x0" ::: "memory");
#endif
#endif
}
__device__ __forceinline__ void wait_tensor0() {
#if HAVE_TDM
#if __has_builtin(__builtin_amdgcn_s_wait_tensorcnt)
  __builtin_amdgcn_s_wait_tensorcnt(0);
#else
  asm volatile("s_wait_tensorcnt 0x0" ::: "memory");
#endif
#endif
}

__device__ __forceinline__ unsigned lds_off_of(const void* p) {
  return (unsigned)(unsigned long long)(AS3 void*)(void*)p;
}

// ---- TDM: 2-D tile load (bf16 elements) global -> LDS, per ISA D# layout ----
__device__ __forceinline__ void tdm_load_2d(unsigned lds_off, const void* gptr,
                                            unsigned td0, unsigned td1,
                                            unsigned tile0, unsigned tile1,
                                            unsigned stride0) {
#if HAVE_TDM
  unsigned long long ga = (unsigned long long)(size_t)gptr;
  v4u g0;
  g0[0] = 1u;                                             // count=1 (valid user D#)
  g0[1] = lds_off;                                        // lds_addr (bytes)
  g0[2] = (unsigned)ga;                                   // global_addr[31:0]
  g0[3] = (unsigned)((ga >> 32) & 0x1FFFFFFull) | (2u << 30); // addr[56:32] | type=2
  v8i g1;
  g1[0] = (int)(1u << 16);                                // data_size=1 -> 2 bytes
  g1[1] = (int)((td0 & 0xFFFFu) << 16);                   // tensor_dim0[15:0]
  g1[2] = (int)(((td0 >> 16) & 0xFFFFu) | ((td1 & 0xFFFFu) << 16));
  g1[3] = (int)(((td1 >> 16) & 0xFFFFu) | ((tile0 & 0xFFFFu) << 16));
  g1[4] = (int)(tile1 & 0xFFFFu);                         // tile_dim1 (tile_dim2=0)
  g1[5] = (int)stride0;                                   // tensor_dim0_stride lo32
  g1[6] = 0; g1[7] = 0;
  v4i z4 = {0, 0, 0, 0};
#if __clang_major__ >= 23
  v8i z8 = {0, 0, 0, 0, 0, 0, 0, 0};
  __builtin_amdgcn_tensor_load_to_lds(g0, g1, z4, z4, z8, 0);
#else
  __builtin_amdgcn_tensor_load_to_lds(g0, g1, z4, z4, 0);
#endif
#endif
}

// ---------------- workspace layout (bytes) ----------------
#define XG_OFF   ((size_t)0)                                  // f32 [BS,G]  512 MB
#define XBF_OFF  (XG_OFF  + (size_t)BS_*G_*4)                 // bf16 [BS,H]  64 MB
#define WIH_OFF  (XBF_OFF + (size_t)BS_*H_*2)                 // bf16 [G,H]    8 MB
#define WHH_OFF  (WIH_OFF + (size_t)G_*H_*2)                  // bf16 [G,H]    8 MB
#define FCW_OFF  (WHH_OFF + (size_t)G_*H_*2)                  // bf16 [H,H]    2 MB
#define SPK_OFF  (FCW_OFF + (size_t)H_*H_*2)                  // bf16 [BS,H]  64 MB
#define HBUF_OFF (SPK_OFF + (size_t)BS_*H_*2)                 // bf16 [2,B,H] 64 KB
#define BIAS_OFF (HBUF_OFF + (size_t)2*B_*H_*2)               // f32  [G]     16 KB

// ---------------- K0: dtype conversion + init ----------------
__global__ __launch_bounds__(256) void k_convert(
    const float* __restrict__ x, const float* __restrict__ Wih,
    const float* __restrict__ Whh, const float* __restrict__ fcw,
    const float* __restrict__ bih, const float* __restrict__ bhh,
    const float* __restrict__ mem,
    __bf16* __restrict__ xbf, __bf16* __restrict__ wihb, __bf16* __restrict__ whhb,
    __bf16* __restrict__ fcwb, float* __restrict__ bias, __bf16* __restrict__ hbuf)
{
  const size_t stride = (size_t)gridDim.x * blockDim.x;
  const size_t i0 = (size_t)blockIdx.x * blockDim.x + threadIdx.x;
  for (size_t i = i0; i < (size_t)BS_*H_; i += stride) xbf[i]  = (__bf16)x[i];
  for (size_t i = i0; i < (size_t)G_*H_;  i += stride) { wihb[i] = (__bf16)Wih[i]; whhb[i] = (__bf16)Whh[i]; }
  for (size_t i = i0; i < (size_t)H_*H_;  i += stride) fcwb[i] = (__bf16)fcw[i];
  for (size_t i = i0; i < (size_t)G_;     i += stride) bias[i] = bih[i] + bhh[i];
  for (size_t i = i0; i < (size_t)B_*H_;  i += stride) hbuf[i] = (__bf16)mem[i];  // h(t=0) buffer 0
}

// ---- shared tile staging for the two big GEMMs (async, double-buffered) ----
__device__ __forceinline__ void stage_AB(const __bf16* __restrict__ A,
                                         const __bf16* __restrict__ W,
                                         __bf16* As, __bf16* Bs,
                                         int tid, int m0, int n0, int k0) {
  int ra = tid >> 1, sega = (tid & 1) * 16;
  const __bf16* ga = &A[(size_t)(m0 + ra)*H_ + k0 + sega];
  __bf16* la = &As[ra*32 + sega];
  cp16_g2l(ga, la);
  cp16_g2l(ga + 8, la + 8);
  int rb = tid >> 2, segb = (tid & 3) * 8;
  cp16_g2l(&W[(size_t)(n0 + rb)*H_ + k0 + segb], &Bs[rb*32 + segb]);
}

// ---------------- K1: xg = Xbf[BS,H] * Wih^T + (b_ih+b_hh) ----------------
__global__ __launch_bounds__(256) void k_gemm_xg(
    const __bf16* __restrict__ A, const __bf16* __restrict__ W,
    const float* __restrict__ bias, float* __restrict__ Cout)
{
  __shared__ __align__(64) __bf16 As[2][128*32];
  __shared__ __align__(64) __bf16 Bs[2][64*32];
  const int tid = threadIdx.x, lane = tid & 31, w = tid >> 5;
  const int m0 = blockIdx.y * 128, n0 = blockIdx.x * 64;
  const int m = lane & 15, half = lane >> 4;
  const int hh8 = half * 8, kh = half * 16;

  stage_AB(A, W, As[0], Bs[0], tid, m0, n0, 0);
  wait_async0();
  __syncthreads();

  v8f acc[4] = {};
  for (int ks = 0; ks < 32; ++ks) {
    const int cur = ks & 1;
    if (ks < 31) stage_AB(A, W, As[cur ^ 1], Bs[cur ^ 1], tid, m0, n0, (ks + 1) * 32);
    v16bf a;
    ((v8bf*)&a)[0] = *(v8bf*)&As[cur][(w*16 + m)*32 + kh];
    ((v8bf*)&a)[1] = *(v8bf*)&As[cur][(w*16 + m)*32 + kh + 8];
#pragma unroll
    for (int ns = 0; ns < 4; ++ns) {
      v16bf b = *(v16bf*)&Bs[cur][(ns*16 + m)*32 + kh];
      acc[ns] = wmma_bf16(a, b, acc[ns]);
    }
    wait_async0();      // next tile landed in LDS
    __syncthreads();
  }
#pragma unroll
  for (int ns = 0; ns < 4; ++ns)
#pragma unroll
    for (int v = 0; v < 8; ++v) {
      int row = m0 + w*16 + v + half*8;
      int col = n0 + ns*16 + m;
      Cout[(size_t)row*G_ + col] = acc[ns][v] + bias[col];
    }
}

// ---------------- K2: sequential SLSTM scan ----------------
// 8 WGs (one cluster) x 8 waves. WG owns 128 hidden units; wave owns 16 and
// chains i/f/g/o WMMA tiles. W_hh tiles are DMA'd by the TDM into
// double-buffered LDS (waves 0-3 each own one gate's descriptor); h(t) is
// exchanged via double-buffered global bf16 + cluster barrier per step.
__global__ __launch_bounds__(256) void k_scan(
    const float* __restrict__ xg, const __bf16* __restrict__ Whh,
    const float* __restrict__ syn, const float* __restrict__ thr_p,
    __bf16* __restrict__ hbuf, __bf16* __restrict__ spks,
    float* __restrict__ c_out, float* __restrict__ h_out)
{
  __shared__ __align__(64) __bf16 Hs[B_*H_];             // 32 KB
#if HAVE_TDM
  __shared__ __align__(64) __bf16 Wbuf[2][4*128*32];     // 64 KB, [buf][gate*128+row][32]
#endif
  const int tid = threadIdx.x, lane = tid & 31, w = tid >> 5;
  const int m = lane & 15, half = lane >> 4;
  const int hh8 = half * 8, kh = half * 16;
  const int jloc = blockIdx.x * 128;          // WG hidden-unit base
  const int colj = jloc + w*16 + m;           // this lane's hidden column
  const float thr = thr_p[0];

  float cst[8], hst[8];
#pragma unroll
  for (int v = 0; v < 8; ++v) {
    int b = v + half*8;
    cst[v] = syn[(size_t)b*H_ + colj];
    hst[v] = 0.f;
  }

  for (int t = 0; t < S_; ++t) {
    // stage h(t) [16,1024] bf16 into LDS via async copies
    const __bf16* hsrc = hbuf + (size_t)(t & 1) * (B_*H_);
#pragma unroll
    for (int u = 0; u < 8; ++u) {
      int e = tid*64 + u*8;
      cp16_g2l(&hsrc[e], &Hs[e]);
    }
#if HAVE_TDM
    if (w < 4) {  // wave g DMAs gate g's 128x32 W_hh tile (chunk 0) into Wbuf[0]
      tdm_load_2d(lds_off_of(&Wbuf[0][w*128*32]),
                  &Whh[(size_t)(w*H_ + jloc) * H_], H_, G_, 32, 128, H_);
    }
#endif
    wait_async0();
#if HAVE_TDM
    if (w < 4) wait_tensor0();
#endif
    __syncthreads();

    // init accumulators from precomputed xg tile (input GEMM + biases)
    v8f acc[4];
#pragma unroll
    for (int g = 0; g < 4; ++g)
#pragma unroll
      for (int v = 0; v < 8; ++v) {
        int b = v + half*8;
        acc[g][v] = xg[((size_t)b*S_ + t)*G_ + g*H_ + colj];
      }

    // gates += h(t) @ W_hh^T, K=1024 in 32-wide chunks
    for (int ks = 0; ks < 32; ++ks) {
      const int k0 = ks * 32;
#if HAVE_TDM
      const int cur = ks & 1;
      if (ks < 31 && w < 4) {  // DMA next chunk while WMMAs run
        tdm_load_2d(lds_off_of(&Wbuf[cur ^ 1][w*128*32]),
                    &Whh[(size_t)(w*H_ + jloc) * H_ + (k0 + 32)], H_, G_, 32, 128, H_);
      }
#endif
      v16bf a;
      ((v8bf*)&a)[0] = *(v8bf*)&Hs[m*H_ + k0 + kh];
      ((v8bf*)&a)[1] = *(v8bf*)&Hs[m*H_ + k0 + kh + 8];
#pragma unroll
      for (int g = 0; g < 4; ++g) {
#if HAVE_TDM
        v16bf b = *(v16bf*)&Wbuf[cur][(g*128 + w*16 + m)*32 + kh];
#else
        const __bf16* wp = &Whh[(size_t)(g*H_ + jloc + w*16 + m)*H_ + k0 + kh];
        __builtin_prefetch(wp + 64, 0, 0);
        v16bf b = *(const v16bf*)wp;
#endif
        acc[g] = wmma_bf16(a, b, acc[g]);
      }
#if HAVE_TDM
      if (ks < 31 && w < 4) wait_tensor0();
      __syncthreads();
#endif
    }

    // LSTM cell update + spike; write h(t+1) bf16 into other buffer
    __bf16* hdst = hbuf + (size_t)((t + 1) & 1) * (B_*H_);
#pragma unroll
    for (int v = 0; v < 8; ++v) {
      int b = v + half*8;
      float ig = sigmoidf_(acc[0][v]);
      float fg = sigmoidf_(acc[1][v]);
      float gg = tanhf(acc[2][v]);
      float og = sigmoidf_(acc[3][v]);
      float c  = fg * cst[v] + ig * gg;
      float h  = og * tanhf(c);
      cst[v] = c; hst[v] = h;
      spks[((size_t)b*S_ + t)*H_ + colj] = (__bf16)(((h - thr) > 0.f) ? 1.0f : 0.0f);
      hdst[(size_t)b*H_ + colj] = (__bf16)h;
    }

    __threadfence();                       // h(t+1) visible device-wide
    __syncthreads();
    __builtin_amdgcn_s_cluster_barrier();  // cluster-wide step handoff
  }

#pragma unroll
  for (int v = 0; v < 8; ++v) {
    int b = v + half*8;
    c_out[(size_t)b*H_ + colj] = cst[v];
    h_out[(size_t)b*H_ + colj] = hst[v];
  }
}

// ---------------- K3: out = DyT( spks @ fc_w^T + fc_b + x ) ----------------
__global__ __launch_bounds__(256) void k_gemm_out(
    const __bf16* __restrict__ A, const __bf16* __restrict__ W,
    const float* __restrict__ fcb, const float* __restrict__ x,
    const float* __restrict__ dyt_a, const float* __restrict__ dyt_w,
    const float* __restrict__ dyt_b, float* __restrict__ out)
{
  __shared__ __align__(64) __bf16 As[2][128*32];
  __shared__ __align__(64) __bf16 Bs[2][64*32];
  const int tid = threadIdx.x, lane = tid & 31, w = tid >> 5;
  const int m0 = blockIdx.y * 128, n0 = blockIdx.x * 64;
  const int m = lane & 15, half = lane >> 4;
  const int hh8 = half * 8, kh = half * 16;
  const float alpha = dyt_a[0];

  stage_AB(A, W, As[0], Bs[0], tid, m0, n0, 0);
  wait_async0();
  __syncthreads();

  v8f acc[4] = {};
  for (int ks = 0; ks < 32; ++ks) {
    const int cur = ks & 1;
    if (ks < 31) stage_AB(A, W, As[cur ^ 1], Bs[cur ^ 1], tid, m0, n0, (ks + 1) * 32);
    v16bf a;
    ((v8bf*)&a)[0] = *(v8bf*)&As[cur][(w*16 + m)*32 + kh];
    ((v8bf*)&a)[1] = *(v8bf*)&As[cur][(w*16 + m)*32 + kh + 8];
#pragma unroll
    for (int ns = 0; ns < 4; ++ns) {
      v16bf b = *(v16bf*)&Bs[cur][(ns*16 + m)*32 + kh];
      acc[ns] = wmma_bf16(a, b, acc[ns]);
    }
    wait_async0();
    __syncthreads();
  }
#pragma unroll
  for (int ns = 0; ns < 4; ++ns)
#pragma unroll
    for (int v = 0; v < 8; ++v) {
      int row = m0 + w*16 + v + half*8;
      int col = n0 + ns*16 + m;
      float z = acc[ns][v] + fcb[col] + x[(size_t)row*H_ + col];
      out[(size_t)row*H_ + col] = dyt_w[col] * tanhf(alpha * z) + dyt_b[col];
    }
}

// ---------------- host launcher ----------------
extern "C" void kernel_launch(void* const* d_in, const int* in_sizes, int n_in,
                              void* d_out, int out_size, void* d_ws, size_t ws_size,
                              hipStream_t stream) {
  const float* x     = (const float*)d_in[0];
  const float* syn   = (const float*)d_in[1];
  const float* mem   = (const float*)d_in[2];
  const float* Wih   = (const float*)d_in[3];
  const float* Whh   = (const float*)d_in[4];
  const float* bih   = (const float*)d_in[5];
  const float* bhh   = (const float*)d_in[6];
  const float* thr   = (const float*)d_in[7];
  /* d_in[8] sdyt_alpha: backward-only */
  const float* fcw   = (const float*)d_in[9];
  const float* fcb   = (const float*)d_in[10];
  const float* dyt_a = (const float*)d_in[11];
  const float* dyt_w = (const float*)d_in[12];
  const float* dyt_b = (const float*)d_in[13];

  char* ws = (char*)d_ws;
  float*  xg   = (float*)(ws + XG_OFF);
  __bf16* xbf  = (__bf16*)(ws + XBF_OFF);
  __bf16* wihb = (__bf16*)(ws + WIH_OFF);
  __bf16* whhb = (__bf16*)(ws + WHH_OFF);
  __bf16* fcwb = (__bf16*)(ws + FCW_OFF);
  __bf16* spkb = (__bf16*)(ws + SPK_OFF);
  __bf16* hbuf = (__bf16*)(ws + HBUF_OFF);
  float*  bias = (float*)(ws + BIAS_OFF);

  float* out   = (float*)d_out;
  float* c_out = out + (size_t)BS_ * H_;
  float* h_out = c_out + (size_t)B_ * H_;

  k_convert<<<2048, 256, 0, stream>>>(x, Wih, Whh, fcw, bih, bhh, mem,
                                      xbf, wihb, whhb, fcwb, bias, hbuf);

  dim3 g1(G_/64, BS_/128);
  k_gemm_xg<<<g1, 256, 0, stream>>>(xbf, wihb, bias, xg);

  k_scan<<<8, 256, 0, stream>>>(xg, whhb, syn, thr, hbuf, spkb, c_out, h_out);

  dim3 g3(H_/64, BS_/128);
  k_gemm_out<<<g3, 256, 0, stream>>>(spkb, fcwb, fcb, x, dyt_a, dyt_w, dyt_b, out);
}
